// LatentUpdate_87935160418331
// MI455X (gfx1250) — matrix operand order repaired and checked by reading, not compiled, not run on hardware
//
#include <hip/hip_runtime.h>
#include <hip/hip_bf16.h>

// ---------------- problem constants ----------------
#define BB   16
#define LL   512
#define KNN  30
#define NN   (BB * LL)          // 8192 nodes
#define EE   (NN * KNN)         // 245760 edges
#define HH   32
#define CC   35                 // fused channels
#define CPAD 36                 // padded channel stride (f32 emb)
#define XPAD 64                 // padded channel stride (f16 xs_rot rows, WMMA-K aligned)
#define HEADS 8
#define ACH  16
#define VCH  8
#define HID  (HEADS * ACH)      // 128
#define INVK 102                // invariant feature length
#define INVP 128                // padded K for WMMA
#define VOUT (HEADS * VCH)      // 64

typedef _Float16 v16h __attribute__((ext_vector_type(16)));
typedef _Float16 h8   __attribute__((ext_vector_type(8)));
typedef float    v8f  __attribute__((ext_vector_type(8)));
typedef unsigned int u32x4 __attribute__((ext_vector_type(4)));
typedef int          i32x4 __attribute__((ext_vector_type(4)));
typedef int          i32x8 __attribute__((ext_vector_type(8)));

#if defined(__has_builtin)
#if __has_builtin(__builtin_amdgcn_tensor_load_to_lds)
#define HAVE_TDM 1
#else
#define HAVE_TDM 0
#endif
#else
#define HAVE_TDM 0
#endif

static __device__ __forceinline__ float silu_f(float x) {
    return x / (1.f + __expf(-x));
}

#if HAVE_TDM
// Tensor Data Mover: DMA a 2-D tile of 2-byte elements (row-major, width = stride)
// from global memory into LDS. D# packing per CDNA5 ISA ch.8 (groups 0/1; rest zero).
// This toolchain exposes the 6-arg builtin: (g0 u32x4, g1 i32x8, i32x4, i32x4, i32x8, cpol).
static __device__ __forceinline__ void tdm_load_2d_f16(unsigned lds_off,
                                                       const void* gptr,
                                                       unsigned width,    // elements per row
                                                       unsigned height) { // rows
    unsigned long long ga = (unsigned long long)(uintptr_t)gptr;
    u32x4 g0;
    g0[0] = 1u;                                             // count=1 (valid, user mode)
    g0[1] = lds_off;                                        // lds_addr (bytes)
    g0[2] = (unsigned)(ga & 0xFFFFFFFFu);                   // global_addr[31:0]
    g0[3] = (unsigned)((ga >> 32) & 0x01FFFFFFu) | (2u << 30); // addr[56:32] | type=2
    i32x8 g1;
    g1[0] = (int)(1u << 16);                                // data_size=1 (2B), wg_mask=0
    g1[1] = (int)((width & 0xFFFFu) << 16);                 // tensor_dim0[15:0]
    g1[2] = (int)((width >> 16) & 0xFFFFu) | (int)((height & 0xFFFFu) << 16);
    g1[3] = (int)((height >> 16) & 0xFFFFu) | (int)((width & 0xFFFFu) << 16); // tile_dim0
    g1[4] = (int)(height & 0xFFFFu);                        // tile_dim1 (tile_dim2=0)
    g1[5] = (int)width;                                     // tensor_dim0_stride[31:0]
    g1[6] = 0;                                              // stride0 hi / stride1 lo
    g1[7] = 0;                                              // stride1 hi
    i32x4 z4 = {0, 0, 0, 0};
    i32x8 z8 = {0, 0, 0, 0, 0, 0, 0, 0};
    __builtin_amdgcn_tensor_load_to_lds(g0, g1, z4, z4, z8, 0);
}
#endif

// A operand (16x32 f16, row-major source with given stride in halves).
// CDNA5 layout: lane<16 -> K {kb..kb+7, kb+16..kb+23}; lane>=16 -> +8.
static __device__ __forceinline__ v16h load_a16(const _Float16* base, int stride,
                                                int lane, int kbase) {
    int m  = lane & 15;
    int ko = kbase + ((lane & 16) ? 8 : 0);
    const h8 lo = *(const h8*)(base + m * stride + ko);
    const h8 hi = *(const h8*)(base + m * stride + ko + 16);
    v16h r;
#pragma unroll
    for (int i = 0; i < 8; ++i) { r[i] = lo[i]; r[i + 8] = hi[i]; }
    return r;
}

// B operand (32x16 f16) from an N-major (transposed) weight buffer Wt[n][k].
// lane<16 -> column n=lane, K {kb..kb+7, kb+16..kb+23} ; lane>=16 -> column n=lane-16, +8.
static __device__ __forceinline__ v16h load_b16(const _Float16* wt, int stride,
                                                int lane, int nbase, int kbase) {
    int n  = nbase + (lane & 15);
    int ks = kbase + ((lane & 16) ? 8 : 0);
    const h8 lo = *(const h8*)(wt + n * stride + ks);
    const h8 hi = *(const h8*)(wt + n * stride + ks + 16);
    v16h r;
#pragma unroll
    for (int i = 0; i < 8; ++i) { r[i] = lo[i]; r[i + 8] = hi[i]; }
    return r;
}

// ---------------- kernel 0: weight transpose/pad to f16 ----------------
__global__ void conv_weights_kernel(const float* __restrict__ W_alpha,  // [102,128]
                                    const float* __restrict__ W_v,      // [35,64]
                                    _Float16* __restrict__ walpha_t,    // [128][128]
                                    _Float16* __restrict__ wv_t) {      // [64][64]
    int tid = blockIdx.x * blockDim.x + threadIdx.x;
    int stride = gridDim.x * blockDim.x;
    for (int i = tid; i < 128 * 128; i += stride) {
        int n = i >> 7, k = i & 127;
        walpha_t[i] = (k < INVK) ? (_Float16)W_alpha[k * 128 + n] : (_Float16)0.f;
    }
    for (int i = tid; i < 64 * 64; i += stride) {
        int n = i >> 6, k = i & 63;
        wv_t[i] = (k < CC) ? (_Float16)W_v[k * 64 + n] : (_Float16)0.f;
    }
}

// ---------------- kernel 1: brute-force per-chain kNN ----------------
__global__ __launch_bounds__(128) void knn_kernel(const float* __restrict__ trans,
                                                  int* __restrict__ src_out) {
    __shared__ float d2s[LL];
    __shared__ float rmin[128];
    __shared__ int   ridx[128];
    int n = blockIdx.x;
    int b = n / LL, l = n % LL;
    int tid = threadIdx.x;
    float cx = trans[n * 3 + 0], cy = trans[n * 3 + 1], cz = trans[n * 3 + 2];
    for (int j = tid; j < LL; j += 128) {
        int m = b * LL + j;
        float dx = trans[m * 3 + 0] - cx;
        float dy = trans[m * 3 + 1] - cy;
        float dz = trans[m * 3 + 2] - cz;
        float d = dx * dx + dy * dy + dz * dz;
        d2s[j] = (j == l) ? 1e9f : d;
    }
    __syncthreads();
    for (int k = 0; k < KNN; ++k) {
        float mv = 1e30f; int mi = 0;
        for (int j = tid; j < LL; j += 128)
            if (d2s[j] < mv) { mv = d2s[j]; mi = j; }
        rmin[tid] = mv; ridx[tid] = mi;
        __syncthreads();
        for (int off = 64; off > 0; off >>= 1) {
            if (tid < off && rmin[tid + off] < rmin[tid]) {
                rmin[tid] = rmin[tid + off]; ridx[tid] = ridx[tid + off];
            }
            __syncthreads();
        }
        if (tid == 0) {
            src_out[n * KNN + k] = b * LL + ridx[0];
            d2s[ridx[0]] = 1e9f;
        }
        __syncthreads();
    }
}

// ---------------- kernel 2: fused node embedding [N][4][36] ----------------
__global__ void build_emb_kernel(const float* __restrict__ rots,
                                 const float* __restrict__ node_emb,
                                 const int* __restrict__ x_mask,
                                 const int* __restrict__ noising_mask,
                                 float* __restrict__ emb) {
    int n = blockIdx.x * blockDim.x + threadIdx.x;
    if (n >= NN) return;
    const float ideal[3][3] = {{-0.525f, 1.363f, 0.f}, {0.f, 0.f, 0.f}, {1.526f, 0.f, 0.f}};
    float R[9];
#pragma unroll
    for (int i = 0; i < 9; ++i) R[i] = rots[n * 9 + i];
    for (int row = 0; row < 4; ++row)
        for (int c = 0; c < HH; ++c)
            emb[((size_t)n * 4 + row) * CPAD + c] = node_emb[((size_t)n * 4 + row) * HH + c];
    int editable = (noising_mask[n] != 0) && (x_mask[n] == 0);
    size_t r0 = (size_t)n * 4 * CPAD;
    emb[r0 + 32] = 0.f; emb[r0 + 33] = 0.f;
    emb[r0 + 34] = editable ? 1.f : 0.f;
    emb[r0 + 35] = 0.f;
#pragma unroll
    for (int i = 0; i < 3; ++i) {
        size_t rr = ((size_t)n * 4 + 1 + i) * CPAD;
#pragma unroll
        for (int a = 0; a < 3; ++a) {
            float fa = R[i * 3 + 0] * ideal[a][0] + R[i * 3 + 1] * ideal[a][1] +
                       R[i * 3 + 2] * ideal[a][2];
            emb[rr + 32 + a] = fa;
        }
        emb[rr + 35] = 0.f;
    }
}

// ---------------- kernel 3: per-edge geometry + feature assembly ----------------
__global__ void edge_prep_kernel(const float* __restrict__ trans,
                                 const int* __restrict__ knn_src,
                                 const float* __restrict__ emb,
                                 _Float16* __restrict__ inv_f16,    // [E][128]
                                 _Float16* __restrict__ xsrot_f16,  // [E*4 (+pad)][64]
                                 float* __restrict__ D1w,           // [E][9]
                                 float* __restrict__ validw) {      // [E]
    int e = blockIdx.x * blockDim.x + threadIdx.x;
    if (e >= EE) return;
    int s = knn_src[e];
    int t = e / KNN;
    float vx = trans[s * 3 + 0] - trans[t * 3 + 0];
    float vy = trans[s * 3 + 1] - trans[t * 3 + 1];
    float vz = trans[s * 3 + 2] - trans[t * 3 + 2];
    float dist = sqrtf(vx * vx + vy * vy + vz * vz);
    int valid = __builtin_isfinite(dist) && (dist > 1e-3f);
    validw[e] = valid ? 1.f : 0.f;

    float inr = 1.f / (dist + 1e-12f);
    float nx0 = vx * inr, nx1 = vy * inr, nx2 = vz * inr;
    float rx, ry, rz;
    if (fabsf(nx0) < 0.9f) { rx = 1.f; ry = 0.f; rz = 0.f; }
    else                   { rx = 0.f; ry = 0.f; rz = 1.f; }
    // nz = normalize(cross(nx, ref))
    float zx = nx1 * rz - nx2 * ry;
    float zy = nx2 * rx - nx0 * rz;
    float zz = nx0 * ry - nx1 * rx;
    float zn = 1.f / (sqrtf(zx * zx + zy * zy + zz * zz) + 1e-12f);
    zx *= zn; zy *= zn; zz *= zn;
    // ny = normalize(cross(nx, nz))
    float yx = nx1 * zz - nx2 * zy;
    float yy = nx2 * zx - nx0 * zz;
    float yz = nx0 * zy - nx1 * zx;
    float yn = 1.f / (sqrtf(yx * yx + yy * yy + yz * yz) + 1e-12f);
    yx *= yn; yy *= yn; yz *= yn;
    // R rows = [nz, nx, ny]; D1[i][j] = R[p(i)][p(j)], p = (1,2,0)
    float R[3][3] = {{zx, zy, zz}, {nx0, nx1, nx2}, {yx, yy, yz}};
    const int p[3] = {1, 2, 0};
    float D[3][3];
#pragma unroll
    for (int i = 0; i < 3; ++i)
#pragma unroll
        for (int j = 0; j < 3; ++j) {
            D[i][j] = R[p[i]][p[j]];
            D1w[(size_t)e * 9 + i * 3 + j] = D[i][j];
        }

    _Float16* invp = inv_f16 + (size_t)e * INVP;
    _Float16* xrp  = xsrot_f16 + (size_t)e * 4 * XPAD;
    const float* es = emb + (size_t)s * 4 * CPAD;
    const float* et = emb + (size_t)t * 4 * CPAD;

    // row 0 (invariant): copy; rows 1..3: rotate by D1
    for (int c = 0; c < CC; ++c) {
        float x0 = es[0 * CPAD + c];
        float x1 = es[1 * CPAD + c];
        float x2 = es[2 * CPAD + c];
        float x3 = es[3 * CPAD + c];
        invp[c]      = (_Float16)x0;
        invp[35 + c] = (_Float16)et[c];
        xrp[0 * XPAD + c] = (_Float16)x0;
#pragma unroll
        for (int i = 0; i < 3; ++i) {
            float xr = D[i][0] * x1 + D[i][1] * x2 + D[i][2] * x3;
            xrp[(1 + i) * XPAD + c] = (_Float16)xr;
        }
    }
    for (int row = 0; row < 4; ++row)
        for (int c = CC; c < XPAD; ++c) xrp[row * XPAD + c] = (_Float16)0.f;

    // edge features: RBF(16) at [70..85], cos(8) at [86..93], sin(8) at [94..101]
    for (int i = 0; i < 16; ++i) {
        float mu = (20.f / 15.f) * (float)i;
        float x = (dist - mu) / 1.25f;
        invp[70 + i] = (_Float16)__expf(-x * x);
    }
    float dd = (float)(s - t);
    for (int j = 0; j < 8; ++j) {
        float freq = __expf(-(float)(2 * j) * 0.575646273f);  // ln(1e4)/16
        float ang = dd * freq;
        invp[86 + j] = (_Float16)__cosf(ang);
        invp[94 + j] = (_Float16)__sinf(ang);
    }
    for (int c = INVK; c < INVP; ++c) invp[c] = (_Float16)0.f;
}

// ---------------- kernel 4: WMMA GEMM inv[E,128] x W_alpha_t -> silu -> logits ----------------
// B matrix (32KB) staged into LDS once per block via the Tensor Data Mover.
__global__ __launch_bounds__(128) void wmma_logits_kernel(const _Float16* __restrict__ inv_f16,
                                                          const _Float16* __restrict__ walpha_t,
                                                          const float* __restrict__ a_vec,
                                                          const float* __restrict__ validw,
                                                          float* __restrict__ logits) {
    __shared__ float lds[4][16 * HID];   // per-wave hidden tile
    __shared__ __attribute__((aligned(16))) _Float16 bLDS[INVP * INVP];  // 32KB weight tile
    int wave = threadIdx.x >> 5;
    int lane = threadIdx.x & 31;
    int tile = blockIdx.x * 4 + wave;    // 16 edges per wave
    const _Float16* A = inv_f16 + (size_t)tile * 16 * INVP;

#if HAVE_TDM
    if (threadIdx.x < 32) {              // wave 0 issues the DMA; EXEC ignored by TDM
        tdm_load_2d_f16((unsigned)(uintptr_t)bLDS, walpha_t, INVP, INVP);
        __builtin_amdgcn_s_wait_tensorcnt(0);
    }
#else
    for (int i = threadIdx.x; i < INVP * INVP; i += 128) bLDS[i] = walpha_t[i];
#endif
    __syncthreads();

    v8f acc[8];
#pragma unroll
    for (int i = 0; i < 8; ++i) acc[i] = 0;

    v16h av[4];
#pragma unroll
    for (int kt = 0; kt < 4; ++kt) av[kt] = load_a16(A, INVP, lane, kt * 32);

#pragma unroll
    for (int nt = 0; nt < 8; ++nt) {
#pragma unroll
        for (int kt = 0; kt < 4; ++kt) {
            v16h bv = load_b16(bLDS, INVP, lane, nt * 16, kt * 32);
            acc[nt] = __builtin_amdgcn_wmma_f32_16x16x32_f16(
                false, av[kt], false, bv, (short)0, acc[nt], false, false);
        }
    }

    int mo = 8 * (lane >> 4);
    int n0 = lane & 15;
#pragma unroll
    for (int nt = 0; nt < 8; ++nt)
#pragma unroll
        for (int r = 0; r < 8; ++r) {
            int row = r + mo;
            int col = nt * 16 + n0;
            lds[wave][row * HID + col] = silu_f(acc[nt][r]);
        }
    __syncthreads();

    // 16 edges x 8 heads = 128 logits per wave; 4 per lane
#pragma unroll
    for (int p = 0; p < 4; ++p) {
        int pair = p * 32 + lane;
        int eL = pair >> 3;
        int head = pair & 7;
        float s = 0.f;
#pragma unroll
        for (int k = 0; k < ACH; ++k)
            s += lds[wave][eL * HID + head * ACH + k] * a_vec[head * ACH + k];
        s = (s > 0.f) ? s : 0.2f * s;           // leaky_relu(0.2)
        int e = tile * 16 + eL;
        if (validw[e] == 0.f) s = -1e9f;
        logits[(size_t)e * HEADS + head] = s;
    }
}

// ---------------- kernel 5: per-node softmax + WMMA value GEMM + rotate-back + FFN ----------------
// A operand read directly from global (node's 120 rows are contiguous, XPAD=64 aligned).
// W_v tile (8KB) staged into LDS via TDM.
__global__ __launch_bounds__(256) void wmma_agg_kernel(const _Float16* __restrict__ xsrot_f16,
                                                       const _Float16* __restrict__ wv_t,
                                                       const float* __restrict__ logits,
                                                       const float* __restrict__ D1w,
                                                       const float* __restrict__ W_o,
                                                       const float* __restrict__ W_g,
                                                       const float* __restrict__ W_2,
                                                       float* __restrict__ out) {
    __shared__ float vbuf[128 * 64];     // alpha-weighted values
    __shared__ __attribute__((aligned(16))) _Float16 bLDS[64 * 64];  // 8KB W_v tile
    __shared__ float alphaS[KNN * HEADS];
    __shared__ float aggS[4 * VOUT];
    __shared__ float yS[4 * HH];
    __shared__ float gS[HH];

    int n = blockIdx.x;
    int tid = threadIdx.x;
    int wave = tid >> 5, lane = tid & 31;
    size_t ebase = (size_t)n * KNN;

#if HAVE_TDM
    if (tid < 32) {
        tdm_load_2d_f16((unsigned)(uintptr_t)bLDS, wv_t, 64, 64);
        __builtin_amdgcn_s_wait_tensorcnt(0);
    }
#else
    for (int i = tid; i < 64 * 64; i += 256) bLDS[i] = wv_t[i];
#endif
    // per-head softmax over 30 contiguous edges
    if (tid < HEADS) {
        int h = tid;
        float mx = -1e30f;
        for (int e = 0; e < KNN; ++e)
            mx = fmaxf(mx, logits[(ebase + e) * HEADS + h]);
        float den = 0.f;
        for (int e = 0; e < KNN; ++e) {
            float ex = __expf(logits[(ebase + e) * HEADS + h] - mx);
            alphaS[e * HEADS + h] = ex;
            den += ex;
        }
        float rinv = 1.f / (den + 1e-9f);
        for (int e = 0; e < KNN; ++e) alphaS[e * HEADS + h] *= rinv;
    }
    __syncthreads();

    // value GEMM: each wave owns one 16-row M tile; K=64 (2 steps), N=64 (4 tiles)
    v8f acc[4];
#pragma unroll
    for (int i = 0; i < 4; ++i) acc[i] = 0;
    const _Float16* Abase = xsrot_f16 + (ebase * 4 + (size_t)wave * 16) * XPAD;
    v16h av[2];
    av[0] = load_a16(Abase, XPAD, lane, 0);
    av[1] = load_a16(Abase, XPAD, lane, 32);
#pragma unroll
    for (int nt = 0; nt < 4; ++nt) {
#pragma unroll
        for (int kt = 0; kt < 2; ++kt) {
            v16h bv = load_b16(bLDS, 64, lane, nt * 16, kt * 32);
            acc[nt] = __builtin_amdgcn_wmma_f32_16x16x32_f16(
                false, av[kt], false, bv, (short)0, acc[nt], false, false);
        }
    }
    // scatter alpha * v  (rows >= 120 gated to zero by alpha)
    int mo = 8 * (lane >> 4);
    int n0 = lane & 15;
#pragma unroll
    for (int nt = 0; nt < 4; ++nt)
#pragma unroll
        for (int r = 0; r < 8; ++r) {
            int row = wave * 16 + r + mo;
            int col = nt * 16 + n0;
            int eL = row >> 2;
            float al = (eL < KNN) ? alphaS[eL * HEADS + (col >> 3)] : 0.f;
            vbuf[row * 64 + col] = acc[nt][r] * al;
        }
    __syncthreads();

    // rotate back (D1^T) + aggregate over 30 edges
    {
        int ro = tid >> 6;     // output row 0..3
        int col = tid & 63;
        float s = 0.f;
        if (ro == 0) {
            for (int e = 0; e < KNN; ++e) s += vbuf[(e * 4) * 64 + col];
        } else {
            int i = ro - 1;
            for (int e = 0; e < KNN; ++e) {
                const float* D = D1w + (ebase + e) * 9;
                s += D[0 * 3 + i] * vbuf[(e * 4 + 1) * 64 + col] +
                     D[1 * 3 + i] * vbuf[(e * 4 + 2) * 64 + col] +
                     D[2 * 3 + i] * vbuf[(e * 4 + 3) * 64 + col];
            }
        }
        aggS[ro * 64 + col] = s;
    }
    __syncthreads();

    // y = agg @ W_o
    if (tid < 128) {
        int ro = tid >> 5, oc = tid & 31;
        float s = 0.f;
        for (int k = 0; k < VOUT; ++k) s += aggS[ro * 64 + k] * W_o[k * HH + oc];
        yS[ro * HH + oc] = s;
    }
    __syncthreads();
    // gate = silu(y0 @ W_g)
    if (tid < HH) {
        float s = 0.f;
        for (int k = 0; k < HH; ++k) s += yS[k] * W_g[k * HH + tid];
        gS[tid] = silu_f(s);
    }
    __syncthreads();
    // out = (y * gate) @ W_2
    if (tid < 128) {
        int ro = tid >> 5, oc = tid & 31;
        float s = 0.f;
        for (int k = 0; k < HH; ++k) s += yS[ro * HH + k] * gS[k] * W_2[k * HH + oc];
        out[((size_t)n * 4 + ro) * HH + oc] = s;
    }
}

// ---------------- host launcher ----------------
static inline size_t align256(size_t x) { return (x + 255) & ~(size_t)255; }

extern "C" void kernel_launch(void* const* d_in, const int* in_sizes, int n_in,
                              void* d_out, int out_size, void* d_ws, size_t ws_size,
                              hipStream_t stream) {
    const float* rots         = (const float*)d_in[0];
    const float* trans        = (const float*)d_in[1];
    const float* node_emb     = (const float*)d_in[2];
    // d_in[3] = batch (unused: chains are contiguous blocks of L)
    const int*   x_mask       = (const int*)d_in[4];
    const int*   noising_mask = (const int*)d_in[5];
    const float* W_alpha      = (const float*)d_in[6];
    const float* a_vec        = (const float*)d_in[7];
    const float* W_v          = (const float*)d_in[8];
    const float* W_o          = (const float*)d_in[9];
    const float* W_g          = (const float*)d_in[10];
    const float* W_2          = (const float*)d_in[11];
    float* out = (float*)d_out;

    char* ws = (char*)d_ws;
    size_t off = 0;
    int*      knn_src  = (int*)(ws + off);      off = align256(off + (size_t)EE * sizeof(int));
    float*    validw   = (float*)(ws + off);    off = align256(off + (size_t)EE * sizeof(float));
    float*    emb      = (float*)(ws + off);    off = align256(off + (size_t)NN * 4 * CPAD * sizeof(float));
    float*    D1w      = (float*)(ws + off);    off = align256(off + (size_t)EE * 9 * sizeof(float));
    float*    logitsW  = (float*)(ws + off);    off = align256(off + (size_t)EE * HEADS * sizeof(float));
    _Float16* inv_f16  = (_Float16*)(ws + off); off = align256(off + (size_t)EE * INVP * sizeof(_Float16));
    _Float16* xsrot    = (_Float16*)(ws + off); off = align256(off + ((size_t)EE * 4 + 16) * XPAD * sizeof(_Float16));
    _Float16* walpha_t = (_Float16*)(ws + off); off = align256(off + (size_t)128 * 128 * sizeof(_Float16));
    _Float16* wv_t     = (_Float16*)(ws + off); off = align256(off + (size_t)64 * 64 * sizeof(_Float16));
    (void)ws_size; (void)in_sizes; (void)n_in; (void)out_size;

    conv_weights_kernel<<<64, 256, 0, stream>>>(W_alpha, W_v, walpha_t, wv_t);
    knn_kernel<<<NN, 128, 0, stream>>>(trans, knn_src);
    build_emb_kernel<<<(NN + 255) / 256, 256, 0, stream>>>(rots, node_emb, x_mask,
                                                           noising_mask, emb);
    edge_prep_kernel<<<(EE + 255) / 256, 256, 0, stream>>>(trans, knn_src, emb,
                                                           inv_f16, xsrot, D1w, validw);
    wmma_logits_kernel<<<EE / 16 / 4, 128, 0, stream>>>(inv_f16, walpha_t, a_vec,
                                                        validw, logitsW);
    wmma_agg_kernel<<<NN, 256, 0, stream>>>(xsrot, wv_t, logitsW, D1w,
                                            W_o, W_g, W_2, out);
}
